// dividedSpaceTimeAttention_74285754351795
// MI455X (gfx1250) — matrix operand order repaired and checked by reading, not compiled, not run on hardware
//
#include <hip/hip_runtime.h>
#include <hip/hip_bf16.h>

// ---------------------------------------------------------------------------
// Divided space-time attention for MI455X (gfx1250, wave32, WMMA).
// Pipeline:
//   1) cvt x -> bf16; transpose+cvt all 5 weight matrices to [N][K] bf16
//   2) 3x WMMA GEMM: yp_{q,v,k} = x @ W{q,k,v} + b   (M=12544,N=2304,K=768)
//   3) fused attention (1 wave per (b,h,n)): S = q k^T (6x wmma), softmax,
//      s1 = (colsum att) @ v  -> g_in [B,196,2304] bf16
//   4) WMMA GEMM: gmat = g_in @ Wt + bt              (M=784, N=768, K=2304)
//   5) 3x WMMA GEMM: yp2 = gmat @ W{q,k,v} + b       (M=784, N=2304,K=768)
//   6) fused attention, mode 1 (rows gathered via (f*196+n)/16) -> s2g
//   7) WMMA GEMM (f32 out): o = s2g @ Wf + bf        (M=784, N=768, K=2304)
//   8) tile o across F frames into d_out
// ---------------------------------------------------------------------------

typedef unsigned short u16;
typedef __bf16  bf16_t;
typedef __attribute__((ext_vector_type(16))) __bf16 v16bf;
typedef __attribute__((ext_vector_type(8)))  float  v8f;

#define H_    12
#define N_    196
#define F_    16
#define DIM_  768
#define D3_   192
#define HD3_  2304
#define B_    4
#define SEQ_  3136            // N_*F_
#define SCALE_ (1.0f/96.0f)   // 1/(sqrt(64)*12)

struct alignas(16) U4 { unsigned x, y, z, w; };
union Frag { v16bf v; U4 q[2]; };

__device__ inline u16 f2bf(float f) {
  unsigned u = __float_as_uint(f);
  unsigned r = u + 0x7FFFu + ((u >> 16) & 1u);   // round-to-nearest-even
  return (u16)(r >> 16);
}
__device__ inline float bf2f(u16 s) { return __uint_as_float(((unsigned)s) << 16); }

// A fragment (16x32 bf16): lane holds row (lane&15), K chunks at kb=(lane>>4)*8
// -> elements 0..7 = K kb..kb+7, elements 8..15 = K kb+16..kb+23  (ISA 7.12.2)
__device__ inline v16bf load_frag_a(const u16* p /*per-lane row base + k*/, int lane) {
  Frag f;
  const u16* pp = p + ((lane >> 4) << 3);
  f.q[0] = *(const U4*)(pp);
  f.q[1] = *(const U4*)(pp + 16);
  return f.v;
}
// B fragment (32x16 bf16) from transposed weights [N][K]: lane holds col (lane&15),
// K chunks at kb=(lane>>4)*8 -> elements 0..7 = K kb..kb+7, 8..15 = K kb+16..kb+23 (2x b128)
__device__ inline v16bf load_frag_b(const u16* p /*per-lane col base + k*/, int lane) {
  Frag f;
  const u16* pp = p + ((lane >> 4) << 3);
  f.q[0] = *(const U4*)(pp);
  f.q[1] = *(const U4*)(pp + 16);
  return f.v;
}

__device__ inline v8f wmma_bf16(v16bf a, v16bf b, v8f c) {
  return __builtin_amdgcn_wmma_f32_16x16x32_bf16(false, a, false, b, (short)0, c,
                                                 false, false);
}

// ---------------------------------------------------------------------------
// Generic WMMA GEMM: C[M,N] = A[M,K](bf16) @ B (stored transposed [N][K] bf16)
// + bias[N].  Block = 8 waves; each wave owns a 32x64 tile (2 M-tiles x 4
// N-tiles = 8 wmma per K-step).  Block tile = 256 rows x 64 cols.
// ---------------------------------------------------------------------------
__global__ __launch_bounds__(256)
void gemm_bf16_wmma(const u16* __restrict__ A, const u16* __restrict__ BT,
                    const float* __restrict__ bias,
                    u16* __restrict__ Cb, float* __restrict__ Cf,
                    int M, int N, int K) {
  const int lane = threadIdx.x & 31;
  const int wave = threadIdx.x >> 5;
  const int lr   = lane & 15;
  const int rowBase = blockIdx.y * 256 + wave * 32;
  const int colBase = blockIdx.x * 64;

  int r0 = rowBase + lr;      if (r0 > M - 1) r0 = M - 1;
  int r1 = rowBase + 16 + lr; if (r1 > M - 1) r1 = M - 1;

  const u16* a0p = A + (size_t)r0 * K;
  const u16* a1p = A + (size_t)r1 * K;
  const u16* bp[4];
#pragma unroll
  for (int nt = 0; nt < 4; ++nt)
    bp[nt] = BT + (size_t)(colBase + nt * 16 + lr) * K;

  v8f acc[2][4];
#pragma unroll
  for (int mi = 0; mi < 2; ++mi)
#pragma unroll
    for (int nt = 0; nt < 4; ++nt)
      acc[mi][nt] = (v8f){0.f,0.f,0.f,0.f,0.f,0.f,0.f,0.f};

  for (int k = 0; k < K; k += 32) {
    // locality 3 -> near-cache (WGP-scope) prefetch of the next A tiles; the
    // weight (B) operand is small and L2/WGP$-resident from reuse.
    __builtin_prefetch(a0p + k + 128, 0, 3);
    __builtin_prefetch(a1p + k + 128, 0, 3);
    v16bf a0 = load_frag_a(a0p + k, lane);
    v16bf a1 = load_frag_a(a1p + k, lane);
#pragma unroll
    for (int nt = 0; nt < 4; ++nt) {
      v16bf b = load_frag_b(bp[nt] + k, lane);
      acc[0][nt] = wmma_bf16(a0, b, acc[0][nt]);
      acc[1][nt] = wmma_bf16(a1, b, acc[1][nt]);
    }
  }

  const int rh = (lane >> 4) << 3;   // C layout: lanes 0-15 rows 0..7, 16-31 rows 8..15
#pragma unroll
  for (int mi = 0; mi < 2; ++mi) {
#pragma unroll
    for (int nt = 0; nt < 4; ++nt) {
      const int c = colBase + nt * 16 + lr;
      const float bv = bias[c];
#pragma unroll
      for (int r = 0; r < 8; ++r) {
        const int row = rowBase + mi * 16 + rh + r;
        if (row < M) {
          const float val = acc[mi][nt][r] + bv;
          if (Cf) Cf[(size_t)row * N + c] = val;
          else    Cb[(size_t)row * N + c] = f2bf(val);
        }
      }
    }
  }
}

// ---------------------------------------------------------------------------
// Fused temporal attention: one wave per (b,h,n).
//   S[f][g] = sum_d q[f][d] k[g][d]         (6x wmma_bf16, K=192)
//   att = softmax(S * SCALE) over g          (shfl_xor row reductions)
//   s1[d] = sum_g (sum_f att[f][g]) v[g][d]  (column-sum trick; 16x192 matvec)
// mode 0: source row(f) = b*3136 + f*196 + n        (pass 1, yp over B*SEQ rows)
// mode 1: source row(f) = b*196 + (f*196+n)/16      (pass 2, yp over B*196 rows)
// ---------------------------------------------------------------------------
__global__ __launch_bounds__(128)
void attn_fused(const u16* __restrict__ Q, const u16* __restrict__ Km,
                const u16* __restrict__ V, u16* __restrict__ Out, int mode) {
  const int lane = threadIdx.x & 31;
  const int idx  = blockIdx.x * 4 + (threadIdx.x >> 5);  // (b*H+h)*N + n, 9408 total
  const int n = idx % N_;
  const int h = (idx / N_) % H_;
  const int b = idx / (N_ * H_);
  const int lr = lane & 15;        // frame index f for A rows / key index j for B cols

  size_t frow = (mode == 0) ? ((size_t)b * SEQ_ + (size_t)lr * N_ + n)
                            : ((size_t)b * N_ + (size_t)(lr * N_ + n) / F_);
  const size_t hoff = (size_t)h * D3_;
  const u16* qp = Q  + frow * HD3_ + hoff;
  const u16* kp = Km + frow * HD3_ + hoff;

  v8f S = (v8f){0.f,0.f,0.f,0.f,0.f,0.f,0.f,0.f};
#pragma unroll
  for (int k = 0; k < D3_; k += 32) {
    v16bf a  = load_frag_a(qp + k, lane);
    v16bf bm = load_frag_b(kp + k, lane);
    S = wmma_bf16(a, bm, S);
  }

  // softmax over g (columns).  Row M = (lane>=16?8:0)+r lives in one 16-lane half
  // per accumulator slot r -> reduce with xor 1,2,4,8.
  float colp = 0.f;
#pragma unroll
  for (int r = 0; r < 8; ++r) {
    float x = S[r] * SCALE_;
    float m = x;
    m = fmaxf(m, __shfl_xor(m, 1, 32));
    m = fmaxf(m, __shfl_xor(m, 2, 32));
    m = fmaxf(m, __shfl_xor(m, 4, 32));
    m = fmaxf(m, __shfl_xor(m, 8, 32));
    float p = __expf(x - m);
    float z = p;
    z += __shfl_xor(z, 1, 32);
    z += __shfl_xor(z, 2, 32);
    z += __shfl_xor(z, 4, 32);
    z += __shfl_xor(z, 8, 32);
    colp += p / z;                       // accumulate column partial of att
  }
  // full column sum over all 16 rows: add the other half-wave's partial
  const float wcol = colp + __shfl_xor(colp, 16, 32);   // lane j & j+16 hold w[j]

  // s1[d] = sum_g w[g] * v[g][d],  d = lane + 32*t
  float acc[6] = {0.f, 0.f, 0.f, 0.f, 0.f, 0.f};
#pragma unroll
  for (int g = 0; g < 16; ++g) {
    const float wg = __shfl(wcol, g, 32);
    const size_t vrow = (mode == 0) ? ((size_t)b * SEQ_ + (size_t)g * N_ + n)
                                    : ((size_t)b * N_ + (size_t)(g * N_ + n) / F_);
    const u16* vp = V + vrow * HD3_ + hoff + lane;
#pragma unroll
    for (int t = 0; t < 6; ++t) acc[t] += wg * bf2f(vp[32 * t]);
  }

  u16* op = Out + ((size_t)b * N_ + n) * HD3_ + hoff + lane;
#pragma unroll
  for (int t = 0; t < 6; ++t) op[32 * t] = f2bf(acc[t]);
}

// --------------------------- small prep/epilogue kernels --------------------
__global__ void cvt_f32_bf16_v8(const float* __restrict__ src,
                                u16* __restrict__ dst, int n8) {
  int t = blockIdx.x * blockDim.x + threadIdx.x;
  if (t >= n8) return;
  const float4* s = (const float4*)src + (size_t)t * 2;
  float4 f0 = s[0], f1 = s[1];
  U4 u;
  u.x = (unsigned)f2bf(f0.x) | ((unsigned)f2bf(f0.y) << 16);
  u.y = (unsigned)f2bf(f0.z) | ((unsigned)f2bf(f0.w) << 16);
  u.z = (unsigned)f2bf(f1.x) | ((unsigned)f2bf(f1.y) << 16);
  u.w = (unsigned)f2bf(f1.z) | ((unsigned)f2bf(f1.w) << 16);
  ((U4*)dst)[t] = u;
}

__global__ void transpose_cvt(const float* __restrict__ src,
                              u16* __restrict__ dst, int R, int C) {
  int t = blockIdx.x * blockDim.x + threadIdx.x;
  if (t >= R * C) return;
  int r = t / C, c = t - r * C;
  dst[(size_t)c * R + r] = f2bf(src[t]);   // dst is [C][R]
}

__global__ void tile_out(const float* __restrict__ o, float* __restrict__ dst,
                         int total4) {
  int t = blockIdx.x * blockDim.x + threadIdx.x;
  if (t >= total4) return;
  int c4 = t % (DIM_ / 4);
  int s  = (t / (DIM_ / 4)) % SEQ_;
  int b  = t / ((DIM_ / 4) * SEQ_);
  int n  = s % N_;
  ((float4*)dst)[t] =
      ((const float4*)o)[((size_t)b * N_ + n) * (DIM_ / 4) + c4];
}

// ---------------------------------------------------------------------------
extern "C" void kernel_launch(void* const* d_in, const int* in_sizes, int n_in,
                              void* d_out, int out_size, void* d_ws, size_t ws_size,
                              hipStream_t stream) {
  (void)in_sizes; (void)n_in; (void)out_size; (void)ws_size;
  const float* x  = (const float*)d_in[0];
  const float* Wq = (const float*)d_in[1];
  const float* bq = (const float*)d_in[2];
  const float* Wk = (const float*)d_in[3];
  const float* bk = (const float*)d_in[4];
  const float* Wv = (const float*)d_in[5];
  const float* bv = (const float*)d_in[6];
  const float* Wt = (const float*)d_in[7];
  const float* bt = (const float*)d_in[8];
  const float* Wf = (const float*)d_in[9];
  const float* bfi = (const float*)d_in[10];
  float* out = (float*)d_out;

  char* wp = (char*)d_ws;
  auto alloc = [&](size_t bytes) -> char* {
    char* p = wp; wp += (bytes + 255) & ~(size_t)255; return p;
  };
  constexpr size_t XB  = (size_t)B_ * SEQ_ * DIM_;   // 9,633,792
  constexpr size_t WSZ = (size_t)DIM_ * HD3_;        // 1,769,472 (all weights)
  constexpr size_t YP  = (size_t)B_ * SEQ_ * HD3_;   // 28,901,376
  constexpr size_t GIN = (size_t)B_ * N_ * HD3_;     // 1,806,336
  constexpr size_t GM  = (size_t)B_ * N_ * DIM_;     // 602,112

  u16* xb   = (u16*)alloc(XB * 2);
  u16* WqT  = (u16*)alloc(WSZ * 2);
  u16* WkT  = (u16*)alloc(WSZ * 2);
  u16* WvT  = (u16*)alloc(WSZ * 2);
  u16* WtT  = (u16*)alloc(WSZ * 2);
  u16* WfT  = (u16*)alloc(WSZ * 2);
  u16* yp_q = (u16*)alloc(YP * 2);
  u16* yp_v = (u16*)alloc(YP * 2);   // value operand comes from Wk (source swap)
  u16* yp_k = (u16*)alloc(YP * 2);   // key   operand comes from Wv
  u16* g_in = (u16*)alloc(GIN * 2);  // s1 rows [B,196,2304]; reused for s2
  u16* gmat = (u16*)alloc(GM * 2);   // mixed rows [B,196,768] bf16
  float* ofin = (float*)alloc(GM * 4);

  // 1) conversions
  cvt_f32_bf16_v8<<<(int)((XB / 8 + 255) / 256), 256, 0, stream>>>(x, xb, (int)(XB / 8));
  const int tgrid = (int)((WSZ + 255) / 256);
  transpose_cvt<<<tgrid, 256, 0, stream>>>(Wq, WqT, DIM_, HD3_);   // -> [2304][768]
  transpose_cvt<<<tgrid, 256, 0, stream>>>(Wk, WkT, DIM_, HD3_);
  transpose_cvt<<<tgrid, 256, 0, stream>>>(Wv, WvT, DIM_, HD3_);
  transpose_cvt<<<tgrid, 256, 0, stream>>>(Wt, WtT, HD3_, DIM_);   // -> [768][2304]
  transpose_cvt<<<tgrid, 256, 0, stream>>>(Wf, WfT, HD3_, DIM_);

  // 2) pass-1 projections (M=12544, N=2304, K=768)
  {
    dim3 g(HD3_ / 64, (B_ * SEQ_) / 256);
    gemm_bf16_wmma<<<g, 256, 0, stream>>>(xb, WqT, bq, yp_q, nullptr,
                                          B_ * SEQ_, HD3_, DIM_);
    gemm_bf16_wmma<<<g, 256, 0, stream>>>(xb, WkT, bk, yp_v, nullptr,
                                          B_ * SEQ_, HD3_, DIM_);
    gemm_bf16_wmma<<<g, 256, 0, stream>>>(xb, WvT, bv, yp_k, nullptr,
                                          B_ * SEQ_, HD3_, DIM_);
  }

  // 3) pass-1 fused attention -> g_in
  attn_fused<<<(B_ * H_ * N_) / 4, 128, 0, stream>>>(yp_q, yp_k, yp_v, g_in, 0);

  // 4) mixing GEMM: gmat = g_in @ Wt + bt  (M=784, N=768, K=2304)
  {
    dim3 g(DIM_ / 64, (B_ * N_ + 255) / 256);
    gemm_bf16_wmma<<<g, 256, 0, stream>>>(g_in, WtT, bt, gmat, nullptr,
                                          B_ * N_, DIM_, HD3_);
  }

  // 5) pass-2 projections on collapsed rows (M=784, N=2304, K=768); reuse yp bufs
  {
    dim3 g(HD3_ / 64, (B_ * N_ + 255) / 256);
    gemm_bf16_wmma<<<g, 256, 0, stream>>>(gmat, WqT, bq, yp_q, nullptr,
                                          B_ * N_, HD3_, DIM_);
    gemm_bf16_wmma<<<g, 256, 0, stream>>>(gmat, WkT, bk, yp_v, nullptr,
                                          B_ * N_, HD3_, DIM_);
    gemm_bf16_wmma<<<g, 256, 0, stream>>>(gmat, WvT, bv, yp_k, nullptr,
                                          B_ * N_, HD3_, DIM_);
  }

  // 6) pass-2 fused attention (gathered rows) -> reuse g_in as s2
  attn_fused<<<(B_ * H_ * N_) / 4, 128, 0, stream>>>(yp_q, yp_k, yp_v, g_in, 1);

  // 7) final GEMM (f32 out): ofin = s2 @ Wf + bf  (M=784, N=768, K=2304)
  {
    dim3 g(DIM_ / 64, (B_ * N_ + 255) / 256);
    gemm_bf16_wmma<<<g, 256, 0, stream>>>(g_in, WfT, bfi, nullptr, ofin,
                                          B_ * N_, DIM_, HD3_);
  }

  // 8) tile across frames into d_out [B, N*F, DIM]
  tile_out<<<(int)((XB / 4 + 255) / 256), 256, 0, stream>>>(ofin, out, (int)(XB / 4));
}
